// UserSubredditSAGE_27049704030344
// MI455X (gfx1250) — matrix operand order repaired and checked by reading, not compiled, not run on hardware
//
#include <hip/hip_runtime.h>
#include <hip/hip_bf16.h>
#include <math.h>

// ---------- WMMA types (CDNA5 / gfx1250, wave32) ----------
typedef __attribute__((ext_vector_type(16))) __bf16 v16bf;
typedef __attribute__((ext_vector_type(8)))  float  v8f;

__device__ inline unsigned short f32_to_bf16u(float f) {
    unsigned u = __float_as_uint(f);
    unsigned rounding = 0x7FFFu + ((u >> 16) & 1u);   // round-to-nearest-even
    return (unsigned short)((u + rounding) >> 16);
}

// monotone float<->uint key for atomicMax on floats
__device__ inline unsigned fkey(float f) {
    unsigned u = __float_as_uint(f);
    return (u & 0x80000000u) ? ~u : (u | 0x80000000u);
}
__device__ inline float fdecode(unsigned k) {
    unsigned u = (k & 0x80000000u) ? (k & 0x7FFFFFFFu) : ~k;
    return __uint_as_float(u);
}

// ---------- conversion kernels ----------
__global__ void cvt_bf16_kernel(const float* __restrict__ in,
                                unsigned short* __restrict__ out, int n) {
    int tid = blockIdx.x * 256 + threadIdx.x;
    if (tid < n) out[tid] = f32_to_bf16u(in[tid]);
}

// W [K,N] f32 row-major -> Wt [N,K] bf16 (so B fragments are contiguous)
__global__ void transpose_cvt_kernel(const float* __restrict__ W,
                                     unsigned short* __restrict__ Wt,
                                     int K, int N) {
    int tid = blockIdx.x * 256 + threadIdx.x;
    if (tid >= K * N) return;
    int k = tid / N, n = tid - k * N;
    Wt[(size_t)n * K + k] = f32_to_bf16u(W[tid]);
}

// ---------- WMMA GEMM: C[M,N] = act(A[M,K] @ B[K,N] + bias) ----------
// A: bf16 row-major [M,K]; Bt: bf16 [N,K] (B transposed).
// One wave computes a 16x64 strip: the A fragment is loaded once per k-step
// and reused across 4 B fragments / 4 WMMA accumulators (4 wmma per 10 b128).
// Layouts used here: A lane(half=lane>>4,r=lane&15): 16 contiguous K at k0+16*half
// of row r (same k order as B). B lane: 16 contiguous K at k0+16*half of
// column r. C/D: lane column n=lane&15, rows i+8*half for VGPR i.
__global__ __launch_bounds__(128) void wmma_gemm_kernel(
    const unsigned short* __restrict__ A,
    const unsigned short* __restrict__ Bt,
    const float* __restrict__ bias,
    float* __restrict__ C,
    int M, int N, int K, int do_relu)
{
    int wave = threadIdx.x >> 5;
    int lane = threadIdx.x & 31;
    int strips_n = N >> 6;                       // 64-wide strips
    int strip = blockIdx.x * 4 + wave;
    if (strip >= (M >> 4) * strips_n) return;
    int mt = strip / strips_n, st = strip - mt * strips_n;
    int half = lane >> 4, r = lane & 15;

    const unsigned short* arow = A + (size_t)(mt * 16 + r) * K;
    const unsigned short* brow0 = Bt + (size_t)(st * 64 + r) * K;

    v8f acc0 = {}, acc1 = {}, acc2 = {}, acc3 = {};
    for (int k0 = 0; k0 < K; k0 += 32) {
        union { uint4 u[2]; v16bf v; } fa, fb0, fb1, fb2, fb3;
        fa.u[0] = *(const uint4*)(arow + k0 + 16 * half);
        fa.u[1] = *(const uint4*)(arow + k0 + 16 * half + 8);
        const unsigned short* bp = brow0 + k0 + 16 * half;
        fb0.u[0] = *(const uint4*)(bp);
        fb0.u[1] = *(const uint4*)(bp + 8);
        fb1.u[0] = *(const uint4*)(bp + (size_t)16 * K);
        fb1.u[1] = *(const uint4*)(bp + (size_t)16 * K + 8);
        fb2.u[0] = *(const uint4*)(bp + (size_t)32 * K);
        fb2.u[1] = *(const uint4*)(bp + (size_t)32 * K + 8);
        fb3.u[0] = *(const uint4*)(bp + (size_t)48 * K);
        fb3.u[1] = *(const uint4*)(bp + (size_t)48 * K + 8);
        acc0 = __builtin_amdgcn_wmma_f32_16x16x32_bf16(false, fa.v, false, fb0.v, (short)0, acc0, false, false);
        acc1 = __builtin_amdgcn_wmma_f32_16x16x32_bf16(false, fa.v, false, fb1.v, (short)0, acc1, false, false);
        acc2 = __builtin_amdgcn_wmma_f32_16x16x32_bf16(false, fa.v, false, fb2.v, (short)0, acc2, false, false);
        acc3 = __builtin_amdgcn_wmma_f32_16x16x32_bf16(false, fa.v, false, fb3.v, (short)0, acc3, false, false);
    }

    float* crow = C + (size_t)(mt * 16 + 8 * half) * N + st * 64 + r;
    #pragma unroll
    for (int t = 0; t < 4; ++t) {
        const v8f& a = (t == 0) ? acc0 : (t == 1) ? acc1 : (t == 2) ? acc2 : acc3;
        float b = bias[st * 64 + t * 16 + r];
        #pragma unroll
        for (int i = 0; i < 8; ++i) {
            float v = a[i] + b;
            if (do_relu) v = fmaxf(v, 0.0f);
            crow[(size_t)i * N + t * 16] = v;
        }
    }
}

// ---------- l2 normalize rows of 128, emit f32 + bf16 ----------
__global__ __launch_bounds__(256) void rownorm_kernel(
    const float* __restrict__ in, float* __restrict__ outf,
    unsigned short* __restrict__ outb, int rows)
{
    int wave = threadIdx.x >> 5, lane = threadIdx.x & 31;
    int row = blockIdx.x * 8 + wave;
    if (row >= rows) return;
    const float* p = in + (size_t)row * 128;
    float v[4], ss = 0.f;
    #pragma unroll
    for (int j = 0; j < 4; ++j) { v[j] = p[lane + 32 * j]; ss += v[j] * v[j]; }
    #pragma unroll
    for (int m = 16; m >= 1; m >>= 1) ss += __shfl_xor(ss, m, 32);
    float scale = 1.0f / fmaxf(sqrtf(ss), 1e-12f);
    #pragma unroll
    for (int j = 0; j < 4; ++j) {
        float o = v[j] * scale;
        size_t idx = (size_t)row * 128 + lane + 32 * j;
        outf[idx] = o;
        if (outb) outb[idx] = f32_to_bf16u(o);
    }
}

// ---------- init per-layer softmax state ----------
__global__ void init_kernel(unsigned* __restrict__ mx, float* __restrict__ den,
                            float* __restrict__ agg, int nmh, int nagg) {
    int tid = blockIdx.x * 256 + threadIdx.x;
    if (tid < nagg) agg[tid] = 0.0f;
    if (tid < nmh) { den[tid] = 0.0f; mx[tid] = 0x007FFFFFu; /* key(-inf) */ }
}

// ---------- edge logits + segment max (wave per edge) ----------
__global__ __launch_bounds__(256) void edge_logits_kernel(
    const float* __restrict__ xl, const float* __restrict__ xr,
    const float* __restrict__ ea, const int* __restrict__ src,
    const int* __restrict__ dst, const float* __restrict__ We,
    const float* __restrict__ att, float* __restrict__ logits,
    unsigned* __restrict__ mxbits, int E)
{
    int wave = threadIdx.x >> 5, lane = threadIdx.x & 31;
    int e = blockIdx.x * 8 + wave;
    if (e >= E) return;
    int s = src[e], d = dst[e];
    float ea0 = ea[2 * e], ea1 = ea[2 * e + 1];
    const float* pl = xl + (size_t)s * 640;
    const float* pr = xr + (size_t)d * 640;
    #pragma unroll
    for (int h = 0; h < 5; ++h) {
        float acc = 0.f;
        #pragma unroll
        for (int j = 0; j < 4; ++j) {
            int idx = h * 128 + lane + 32 * j;
            float m = pl[idx] + pr[idx] + ea0 * We[idx] + ea1 * We[640 + idx];
            m = (m > 0.f) ? m : 0.2f * m;          // leaky_relu(0.2)
            acc += m * att[idx];                    // att flattened [H*128]
        }
        #pragma unroll
        for (int mk = 16; mk >= 1; mk >>= 1) acc += __shfl_xor(acc, mk, 32);
        if (lane == 0) {
            logits[(size_t)e * 5 + h] = acc;
            atomicMax(&mxbits[(size_t)d * 5 + h], fkey(acc));
        }
    }
}

// ---------- exp + segment denominator (thread per edge-head) ----------
__global__ void edge_exp_kernel(float* __restrict__ logits,
                                const int* __restrict__ dst,
                                const unsigned* __restrict__ mxbits,
                                float* __restrict__ den, int EH)
{
    int tid = blockIdx.x * 256 + threadIdx.x;
    if (tid >= EH) return;
    int e = tid / 5, h = tid - e * 5;
    int d = dst[e];
    float mx = fdecode(mxbits[(size_t)d * 5 + h]);
    float ex = __expf(logits[tid] - mx);
    logits[tid] = ex;                                // in-place: logits -> ex
    atomicAdd(&den[(size_t)d * 5 + h], ex);
}

// ---------- alpha-weighted scatter aggregation (wave per edge) ----------
__global__ __launch_bounds__(256) void edge_aggregate_kernel(
    const float* __restrict__ xl, const float* __restrict__ exv,
    const float* __restrict__ den, const int* __restrict__ src,
    const int* __restrict__ dst, float* __restrict__ agg, int E)
{
    int wave = threadIdx.x >> 5, lane = threadIdx.x & 31;
    int e = blockIdx.x * 8 + wave;
    if (e >= E) return;
    int s = src[e], d = dst[e];
    const float* pl = xl + (size_t)s * 640;
    float* pd = agg + (size_t)d * 640;
    #pragma unroll
    for (int h = 0; h < 5; ++h) {
        float alpha = exv[(size_t)e * 5 + h] /
                      (den[(size_t)d * 5 + h] + 1e-16f);
        #pragma unroll
        for (int j = 0; j < 4; ++j) {
            int idx = h * 128 + lane + 32 * j;
            atomicAdd(&pd[idx], pl[idx] * alpha);
        }
    }
}

// ---------- layer-1 epilogue: head mean + bias + residual + relu ----------
__global__ void post1_kernel(const float* __restrict__ agg,
                             const float* __restrict__ bias,
                             const float* __restrict__ u,
                             float* __restrict__ u1f,
                             unsigned short* __restrict__ u1b, int total)
{
    int tid = blockIdx.x * 256 + threadIdx.x;
    if (tid >= total) return;
    int n = tid >> 7, c = tid & 127;
    float sum = 0.f;
    #pragma unroll
    for (int h = 0; h < 5; ++h) sum += agg[(size_t)n * 640 + h * 128 + c];
    float v = fmaxf(sum * 0.2f + bias[c] + u[tid], 0.0f);
    u1f[tid] = v;
    u1b[tid] = f32_to_bf16u(v);
}

// ---------- layer-2 epilogue: head mean + bias + residual + l2norm ----------
__global__ __launch_bounds__(256) void post2_kernel(
    const float* __restrict__ agg, const float* __restrict__ bias,
    const float* __restrict__ u1, float* __restrict__ out, int rows)
{
    int wave = threadIdx.x >> 5, lane = threadIdx.x & 31;
    int n = blockIdx.x * 8 + wave;
    if (n >= rows) return;
    float v[4], ss = 0.f;
    #pragma unroll
    for (int j = 0; j < 4; ++j) {
        int c = lane + 32 * j;
        float sum = 0.f;
        #pragma unroll
        for (int h = 0; h < 5; ++h) sum += agg[(size_t)n * 640 + h * 128 + c];
        v[j] = sum * 0.2f + bias[c] + u1[(size_t)n * 128 + c];
        ss += v[j] * v[j];
    }
    #pragma unroll
    for (int m = 16; m >= 1; m >>= 1) ss += __shfl_xor(ss, m, 32);
    float scale = 1.0f / fmaxf(sqrtf(ss), 1e-12f);
    #pragma unroll
    for (int j = 0; j < 4; ++j)
        out[(size_t)n * 128 + lane + 32 * j] = v[j] * scale;
}

// =====================================================================
extern "C" void kernel_launch(void* const* d_in, const int* in_sizes, int n_in,
                              void* d_out, int out_size, void* d_ws, size_t ws_size,
                              hipStream_t stream)
{
    const float* x_sub  = (const float*)d_in[0];
    const float* x_user = (const float*)d_in[1];
    const float* e_attr = (const float*)d_in[2];
    const float* W_sub  = (const float*)d_in[3];
    const float* b_sub  = (const float*)d_in[4];
    const float* W_user = (const float*)d_in[5];
    const float* b_user = (const float*)d_in[6];
    const float* Wl1 = (const float*)d_in[7];  const float* bl1 = (const float*)d_in[8];
    const float* Wr1 = (const float*)d_in[9];  const float* br1 = (const float*)d_in[10];
    const float* We1 = (const float*)d_in[11]; const float* att1 = (const float*)d_in[12];
    const float* bias1 = (const float*)d_in[13];
    const float* Wl2 = (const float*)d_in[14]; const float* bl2 = (const float*)d_in[15];
    const float* Wr2 = (const float*)d_in[16]; const float* br2 = (const float*)d_in[17];
    const float* We2 = (const float*)d_in[18]; const float* att2 = (const float*)d_in[19];
    const float* bias2 = (const float*)d_in[20];
    const int* e_src = (const int*)d_in[21];
    const int* e_dst = (const int*)d_in[22];

    const int D  = 256, Hd = 128, H = 5, HC = 640;
    const int N_sub  = in_sizes[0] / D;
    const int N_user = in_sizes[1] / D;
    const int E      = in_sizes[21];

    // ---- workspace bump allocator ----
    char* wp = (char*)d_ws;
    auto alloc = [&](size_t bytes) -> char* {
        char* r = wp; wp += (bytes + 255) & ~(size_t)255; return r;
    };
    unsigned short* xsub_b  = (unsigned short*)alloc((size_t)N_sub  * D * 2);
    unsigned short* xuser_b = (unsigned short*)alloc((size_t)N_user * D * 2);
    unsigned short* Wsub_t  = (unsigned short*)alloc((size_t)D * Hd * 2);
    unsigned short* Wuser_t = (unsigned short*)alloc((size_t)D * Hd * 2);
    unsigned short* Wl1_t   = (unsigned short*)alloc((size_t)Hd * HC * 2);
    unsigned short* Wr1_t   = (unsigned short*)alloc((size_t)Hd * HC * 2);
    unsigned short* Wl2_t   = (unsigned short*)alloc((size_t)Hd * HC * 2);
    unsigned short* Wr2_t   = (unsigned short*)alloc((size_t)Hd * HC * 2);
    unsigned short* s_b     = (unsigned short*)alloc((size_t)N_sub  * Hd * 2);
    float*          u_f     = (float*)         alloc((size_t)N_user * Hd * 4);
    unsigned short* u_b     = (unsigned short*)alloc((size_t)N_user * Hd * 2);
    float*          xl      = (float*)         alloc((size_t)N_sub  * HC * 4); // also s_pre
    float*          xr      = (float*)         alloc((size_t)N_user * HC * 4); // also u_pre
    float*          logits  = (float*)         alloc((size_t)E * H * 4);
    unsigned*       mxbits  = (unsigned*)      alloc((size_t)N_user * H * 4);
    float*          den     = (float*)         alloc((size_t)N_user * H * 4);
    float*          agg     = (float*)         alloc((size_t)N_user * HC * 4);
    float*          u1_f    = (float*)         alloc((size_t)N_user * Hd * 4);
    unsigned short* u1_b    = (unsigned short*)alloc((size_t)N_user * Hd * 2);

    float* out_s  = (float*)d_out;                       // [N_sub, 128]
    float* out_u2 = out_s + (size_t)N_sub * Hd;          // [N_user, 128]

    auto cdiv = [](long long a, long long b) { return (int)((a + b - 1) / b); };

    // ---- 1. conversions ----
    cvt_bf16_kernel<<<cdiv((size_t)N_sub * D, 256), 256, 0, stream>>>(x_sub, xsub_b, N_sub * D);
    cvt_bf16_kernel<<<cdiv((size_t)N_user * D, 256), 256, 0, stream>>>(x_user, xuser_b, N_user * D);
    transpose_cvt_kernel<<<cdiv(D * Hd, 256), 256, 0, stream>>>(W_sub,  Wsub_t,  D, Hd);
    transpose_cvt_kernel<<<cdiv(D * Hd, 256), 256, 0, stream>>>(W_user, Wuser_t, D, Hd);
    transpose_cvt_kernel<<<cdiv(Hd * HC, 256), 256, 0, stream>>>(Wl1, Wl1_t, Hd, HC);
    transpose_cvt_kernel<<<cdiv(Hd * HC, 256), 256, 0, stream>>>(Wr1, Wr1_t, Hd, HC);
    transpose_cvt_kernel<<<cdiv(Hd * HC, 256), 256, 0, stream>>>(Wl2, Wl2_t, Hd, HC);
    transpose_cvt_kernel<<<cdiv(Hd * HC, 256), 256, 0, stream>>>(Wr2, Wr2_t, Hd, HC);

    // ---- 2. projections + l2norm (s -> d_out, bf16 copy; u -> ws) ----
    {
        int strips = (N_sub / 16) * (Hd / 64);
        wmma_gemm_kernel<<<cdiv(strips, 4), 128, 0, stream>>>(xsub_b, Wsub_t, b_sub, xl, N_sub, Hd, D, 1);
        rownorm_kernel<<<cdiv(N_sub, 8), 256, 0, stream>>>(xl, out_s, s_b, N_sub);
        strips = (N_user / 16) * (Hd / 64);
        wmma_gemm_kernel<<<cdiv(strips, 4), 128, 0, stream>>>(xuser_b, Wuser_t, b_user, xr, N_user, Hd, D, 1);
        rownorm_kernel<<<cdiv(N_user, 8), 256, 0, stream>>>(xr, u_f, u_b, N_user);
    }

    int strips_l = (N_sub / 16) * (HC / 64);
    int strips_r = (N_user / 16) * (HC / 64);
    int ninit = N_user * HC;

    // ---- 3. GAT layer 1 ----
    wmma_gemm_kernel<<<cdiv(strips_l, 4), 128, 0, stream>>>(s_b, Wl1_t, bl1, xl, N_sub, HC, Hd, 0);
    wmma_gemm_kernel<<<cdiv(strips_r, 4), 128, 0, stream>>>(u_b, Wr1_t, br1, xr, N_user, HC, Hd, 0);
    init_kernel<<<cdiv(ninit, 256), 256, 0, stream>>>(mxbits, den, agg, N_user * H, ninit);
    edge_logits_kernel<<<cdiv(E, 8), 256, 0, stream>>>(xl, xr, e_attr, e_src, e_dst, We1, att1, logits, mxbits, E);
    edge_exp_kernel<<<cdiv((size_t)E * H, 256), 256, 0, stream>>>(logits, e_dst, mxbits, den, E * H);
    edge_aggregate_kernel<<<cdiv(E, 8), 256, 0, stream>>>(xl, logits, den, e_src, e_dst, agg, E);
    post1_kernel<<<cdiv((size_t)N_user * Hd, 256), 256, 0, stream>>>(agg, bias1, u_f, u1_f, u1_b, N_user * Hd);

    // ---- 4. GAT layer 2 ----
    wmma_gemm_kernel<<<cdiv(strips_l, 4), 128, 0, stream>>>(s_b, Wl2_t, bl2, xl, N_sub, HC, Hd, 0);
    wmma_gemm_kernel<<<cdiv(strips_r, 4), 128, 0, stream>>>(u1_b, Wr2_t, br2, xr, N_user, HC, Hd, 0);
    init_kernel<<<cdiv(ninit, 256), 256, 0, stream>>>(mxbits, den, agg, N_user * H, ninit);
    edge_logits_kernel<<<cdiv(E, 8), 256, 0, stream>>>(xl, xr, e_attr, e_src, e_dst, We2, att2, logits, mxbits, E);
    edge_exp_kernel<<<cdiv((size_t)E * H, 256), 256, 0, stream>>>(logits, e_dst, mxbits, den, E * H);
    edge_aggregate_kernel<<<cdiv(E, 8), 256, 0, stream>>>(logits == nullptr ? xl : xl, logits, den, e_src, e_dst, agg, E);
    post2_kernel<<<cdiv(N_user, 8), 256, 0, stream>>>(agg, bias2, u1_f, out_u2, N_user);
}
